// Attention_91087666414100
// MI455X (gfx1250) — matrix operand order repaired and checked by reading, not compiled, not run on hardware
//
#include <hip/hip_runtime.h>
#include <hip/hip_bf16.h>

typedef __attribute__((ext_vector_type(16))) _Float16 v16h;
typedef __attribute__((ext_vector_type(8)))  _Float16 v8h;
typedef __attribute__((ext_vector_type(8)))  float    v8f;

#define HEADS    12
#define DIM_HEAD 64
#define B_       8
#define F_       8
#define NSP      196            // tokens per frame
#define NTOT     1569           // 1 + F_*NSP
#define DIM      768
#define BH       (B_ * HEADS)   // 96
#define M_ROWS   (B_ * NTOT)    // 12552
#define QKV_N    (3 * HEADS * DIM_HEAD) // 2304

union hv16 { v16h v; v8h p[2]; _Float16 h[16]; };

// Async 16B global->LDS copy (CDNA5 async engine, tracked by ASYNCcnt).
// lds_off: wave-relative LDS byte offset (low 32 bits of a flat LDS pointer).
__device__ __forceinline__ void async_copy_b128(unsigned lds_off, const void* gaddr) {
    asm volatile("global_load_async_to_lds_b128 %0, %1, off"
                 :: "v"(lds_off), "v"(gaddr) : "memory");
}
__device__ __forceinline__ void async_wait0() {
    asm volatile("s_wait_asynccnt 0x0" ::: "memory");
}
__device__ __forceinline__ unsigned lds_off_of(const void* p) {
    return (unsigned)(unsigned long long)p;   // flat LDS addr low 32 = LDS offset
}

// ---------------------------------------------------------------------------
// WMMA GEMM: C(MxN) = A(MxK, AT=float|f16) * B(KxN fp32), f16 staged in LDS,
// f32 accumulation via v_wmma_f32_16x16x32_f16.
//   A LDS: row-major [64][32]; AT=float -> b128 loads + cvt_pk; AT=f16 ->
//          two global_load_async_to_lds_b128 per thread (pure DMA).
//   B LDS: fragment layout [nTile(4)][kGrp(2)][col(16)][k(16)] -> B-frag =
//          one aligned 32B chunk per lane (2x ds_load_b128).
// EPI==0: scatter qkv into head-major f16 q/k/v buffers (q scaled by 1/8).
// EPI==1: C = A*B + bias, fp32.
// Block: 128 threads (4 waves, 2x2 wave grid), tile 64x64, K-step 32.
// ---------------------------------------------------------------------------
template<int EPI, typename AT>
__global__ __launch_bounds__(128)
void wmma_gemm_kernel(const AT* __restrict__ A, const float* __restrict__ Bm,
                      int M, int K, int Ncols,
                      _Float16* __restrict__ qb, _Float16* __restrict__ kb,
                      _Float16* __restrict__ vb,
                      float* __restrict__ Cout, const float* __restrict__ bias)
{
    __shared__ __attribute__((aligned(32))) _Float16 Ash[64][32];
    __shared__ __attribute__((aligned(32))) _Float16 Bsh[64 * 32];

    const int tid   = threadIdx.x;
    const int lane  = tid & 31;
    const int wave  = tid >> 5;
    const int warpM = wave >> 1;          // 0..1
    const int warpN = wave & 1;           // 0..1
    const int tileM = blockIdx.x * 64;
    const int tileN = blockIdx.y * 64;

    v8f acc[2][2] = {};

    const int lr  = lane & 15;
    const int kb8 = (lane >> 4) * 16;     // A-frag K base (contiguous layout)

    // A staging mapping: row = tid/2, 16 consecutive K per thread
    const int  ar     = tid >> 1;
    const int  acb    = (tid & 1) * 16;
    const int  agrow  = tileM + ar;
    const bool avalid = agrow < M;
    const AT*  asrc   = A + (long)agrow * K + acb;
    const unsigned aldsoff = lds_off_of(&Ash[ar][acb]);

    // B staging mapping: column per thread (lane-coalesced), 16 consecutive K
    const int  bn   = tid & 63;
    const int  bkg  = (tid >> 6) * 16;    // 0 or 16
    const float* bsrc = Bm + (long)bkg * Ncols + tileN + bn;
    _Float16* bdst = &Bsh[((bn >> 4) * 2 + (bkg >> 4)) * 256 + (bn & 15) * 16];

    for (int k0 = 0; k0 < K; k0 += 32) {
        // ---- stage A ----
        if constexpr (sizeof(AT) == 4) {   // fp32 -> f16 convert path
            hv16 aval;
            if (avalid) {
                const float4* s4 = (const float4*)(asrc + k0);
                #pragma unroll
                for (int c = 0; c < 4; ++c) {
                    float4 f = s4[c];
                    aval.h[c * 4 + 0] = (_Float16)f.x;
                    aval.h[c * 4 + 1] = (_Float16)f.y;
                    aval.h[c * 4 + 2] = (_Float16)f.z;
                    aval.h[c * 4 + 3] = (_Float16)f.w;
                }
            } else {
                #pragma unroll
                for (int i = 0; i < 16; ++i) aval.h[i] = (_Float16)0.f;
            }
            *(v16h*)&Ash[ar][acb] = aval.v;
        } else {                            // f16 A: async DMA straight to LDS
            if (avalid) {
                const char* g = (const char*)(asrc + k0);
                async_copy_b128(aldsoff,      g);
                async_copy_b128(aldsoff + 16, g + 16);
            } else {
                hv16 z;
                #pragma unroll
                for (int i = 0; i < 16; ++i) z.h[i] = (_Float16)0.f;
                *(v16h*)&Ash[ar][acb] = z.v;
            }
        }

        // ---- stage B (16 lane-coalesced loads, one 32B LDS store) ----
        {
            hv16 bval;
            const float* bs = bsrc + (long)k0 * Ncols;
            #pragma unroll
            for (int j = 0; j < 16; ++j)
                bval.h[j] = (_Float16)bs[(long)j * Ncols];
            *(v16h*)bdst = bval.v;
        }

        // prefetch next K-slice into WGP-level caches
        if (k0 + 32 < K) {
            if (avalid) __builtin_prefetch(asrc + k0 + 32, 0, 3);
            __builtin_prefetch(bsrc + (long)(k0 + 32) * Ncols, 0, 3);
        }
        if constexpr (sizeof(AT) == 2) async_wait0();
        __syncthreads();

        hv16 afrag[2], bfrag[2];
        #pragma unroll
        for (int i = 0; i < 2; ++i) {
            const int m0 = warpM * 32 + i * 16 + lr;
            afrag[i].p[0] = *(const v8h*)&Ash[m0][kb8];
            afrag[i].p[1] = *(const v8h*)&Ash[m0][kb8 + 8];
        }
        #pragma unroll
        for (int j = 0; j < 2; ++j)
            bfrag[j].v = *(const v16h*)
                &Bsh[((warpN * 2 + j) * 2 + (lane >> 4)) * 256 + (lane & 15) * 16];

        #pragma unroll
        for (int i = 0; i < 2; ++i)
            #pragma unroll
            for (int j = 0; j < 2; ++j)
                acc[i][j] = __builtin_amdgcn_wmma_f32_16x16x32_f16(
                    false, afrag[i].v, false, bfrag[j].v,
                    (short)0, acc[i][j], false, false);
        __syncthreads();
    }

    // Epilogue: C layout row = v + 8*(lane>>4), col = lane&15
    #pragma unroll
    for (int i = 0; i < 2; ++i) {
        #pragma unroll
        for (int j = 0; j < 2; ++j) {
            #pragma unroll
            for (int v = 0; v < 8; ++v) {
                int row = tileM + warpM * 32 + i * 16 + v + ((lane >> 4) * 8);
                int col = tileN + warpN * 32 + j * 16 + (lane & 15);
                if (row >= M) continue;
                float val = acc[i][j][v];
                if (EPI == 0) {
                    int which = col / DIM;            // 0=q, 1=k, 2=v
                    int rem   = col - which * DIM;
                    int hh    = rem >> 6;
                    int dd    = rem & 63;
                    int bb    = row / NTOT;
                    int tt    = row - bb * NTOT;
                    long oidx = (((long)(bb * HEADS + hh)) * NTOT + tt) * DIM_HEAD + dd;
                    if (which == 0)      qb[oidx] = (_Float16)(val * 0.125f); // q*SCALE
                    else if (which == 1) kb[oidx] = (_Float16)val;
                    else                 vb[oidx] = (_Float16)val;
                } else {
                    Cout[(long)row * Ncols + col] = val + bias[col];
                }
            }
        }
    }
}

// ---------------------------------------------------------------------------
// cls-token attention: one block per head (bh=96). Exact fp32 softmax over
// all 1569 keys; writes cls_attn output and the cls row of attnout (f16).
// ---------------------------------------------------------------------------
__global__ __launch_bounds__(256)
void cls_attn_kernel(const _Float16* __restrict__ qb, const _Float16* __restrict__ kb,
                     const _Float16* __restrict__ vb,
                     float* __restrict__ cls_attn_out, _Float16* __restrict__ attnout)
{
    __shared__ float qsh[64];
    __shared__ float simsh[NTOT];
    __shared__ float red[256];

    const int H   = blockIdx.x;
    const int tid = threadIdx.x;

    if (tid < 64) qsh[tid] = (float)qb[((long)H * NTOT) * 64 + tid];
    __syncthreads();

    float lmax = -1e30f;
    for (int t = tid; t < NTOT; t += 256) {
        const _Float16* kr = kb + ((long)H * NTOT + t) * 64;
        float s = 0.f;
        #pragma unroll
        for (int c = 0; c < 4; ++c) {
            hv16 kf; kf.v = *(const v16h*)&kr[c * 16];
            #pragma unroll
            for (int d = 0; d < 16; ++d) s += qsh[c * 16 + d] * (float)kf.h[d];
        }
        simsh[t] = s;
        lmax = fmaxf(lmax, s);
    }
    red[tid] = lmax; __syncthreads();
    for (int st = 128; st > 0; st >>= 1) {
        if (tid < st) red[tid] = fmaxf(red[tid], red[tid + st]);
        __syncthreads();
    }
    float mx = red[0]; __syncthreads();

    float lsum = 0.f;
    for (int t = tid; t < NTOT; t += 256) {
        float e = __expf(simsh[t] - mx);
        simsh[t] = e; lsum += e;
    }
    red[tid] = lsum; __syncthreads();
    for (int st = 128; st > 0; st >>= 1) {
        if (tid < st) red[tid] += red[tid + st];
        __syncthreads();
    }
    float inv = 1.f / red[0]; __syncthreads();

    for (int t = tid; t < NTOT; t += 256) {
        float a = simsh[t] * inv;
        simsh[t] = a;
        cls_attn_out[(long)H * NTOT + t] = a;
    }
    __syncthreads();

    const int d = tid & 63, grp = tid >> 6;        // 4 partial groups
    float accv = 0.f;
    for (int t = grp; t < NTOT; t += 4)
        accv += simsh[t] * (float)vb[((long)H * NTOT + t) * 64 + d];
    red[tid] = accv; __syncthreads();
    if (grp == 0) {
        float o = red[d] + red[64 + d] + red[128 + d] + red[192 + d];
        int bb = H / HEADS, hh = H - bb * HEADS;
        attnout[((long)bb * NTOT) * DIM + hh * 64 + d] = (_Float16)o;   // t = 0
    }
}

// ---------------------------------------------------------------------------
// Frame-local attention: one block per (head, frame) group (768 blocks).
// 197 keys (cls + 196) staged into LDS via async global->LDS DMA; one query
// row per thread, 2-pass fp32 softmax. Output written f16.
// ---------------------------------------------------------------------------
__global__ __launch_bounds__(256)
void frame_attn_kernel(const _Float16* __restrict__ qb, const _Float16* __restrict__ kb,
                       const _Float16* __restrict__ vb, _Float16* __restrict__ attnout)
{
    __shared__ __attribute__((aligned(32))) _Float16 Ksh[197 * 64];
    __shared__ __attribute__((aligned(32))) _Float16 Vsh[197 * 64];

    const int g   = blockIdx.x;      // H * F_ + fr
    const int H   = g >> 3;
    const int fr  = g & 7;
    const int tid = threadIdx.x;

    // stage K/V rows as 16B async DMA chunks (global_load_async_to_lds_b128)
    for (int c = tid; c < 197 * 8; c += 256) {
        int j = c >> 3, part = (c & 7) * 8;
        int tok = (j == 0) ? 0 : (1 + fr * NSP + (j - 1));
        long src = ((long)H * NTOT + tok) * 64 + part;
        async_copy_b128(lds_off_of(&Ksh[j * 64 + part]), &kb[src]);
        async_copy_b128(lds_off_of(&Vsh[j * 64 + part]), &vb[src]);
    }
    async_wait0();
    __syncthreads();

    const int m = tid;
    if (m >= NSP) return;
    const int t = 1 + fr * NSP + m;

    float qv[64];
    {
        const _Float16* qr = qb + ((long)H * NTOT + t) * 64;
        #pragma unroll
        for (int c = 0; c < 4; ++c) {
            hv16 qf; qf.v = *(const v16h*)&qr[c * 16];
            #pragma unroll
            for (int d = 0; d < 16; ++d) qv[c * 16 + d] = (float)qf.h[d];
        }
    }

    float mx = -1e30f;
    for (int j = 0; j < 197; ++j) {
        float s = 0.f;
        #pragma unroll
        for (int c = 0; c < 4; ++c) {
            hv16 kf; kf.v = *(const v16h*)&Ksh[j * 64 + c * 16];
            #pragma unroll
            for (int d = 0; d < 16; ++d) s += qv[c * 16 + d] * (float)kf.h[d];
        }
        mx = fmaxf(mx, s);
    }

    float acc[64];
    #pragma unroll
    for (int d = 0; d < 64; ++d) acc[d] = 0.f;
    float l = 0.f;
    for (int j = 0; j < 197; ++j) {
        float s = 0.f;
        #pragma unroll
        for (int c = 0; c < 4; ++c) {
            hv16 kf; kf.v = *(const v16h*)&Ksh[j * 64 + c * 16];
            #pragma unroll
            for (int d = 0; d < 16; ++d) s += qv[c * 16 + d] * (float)kf.h[d];
        }
        float p = __expf(s - mx);
        l += p;
        #pragma unroll
        for (int c = 0; c < 4; ++c) {
            hv16 vf; vf.v = *(const v16h*)&Vsh[j * 64 + c * 16];
            #pragma unroll
            for (int d = 0; d < 16; ++d) acc[c * 16 + d] += p * (float)vf.h[d];
        }
    }

    const float inv = 1.f / l;
    const int bb = H / HEADS, hh = H - bb * HEADS;
    _Float16* orow = attnout + ((long)bb * NTOT + t) * DIM + hh * 64;
    #pragma unroll
    for (int c = 0; c < 4; ++c) {
        hv16 o;
        #pragma unroll
        for (int d = 0; d < 16; ++d) o.h[d] = (_Float16)(acc[c * 16 + d] * inv);
        *(v16h*)&orow[c * 16] = o.v;
    }
}

// ---------------------------------------------------------------------------
extern "C" void kernel_launch(void* const* d_in, const int* in_sizes, int n_in,
                              void* d_out, int out_size, void* d_ws, size_t ws_size,
                              hipStream_t stream) {
    const float* x     = (const float*)d_in[0];   // (8, 1569, 768)
    const float* Wqkv  = (const float*)d_in[1];   // (768, 2304)
    const float* Wout  = (const float*)d_in[2];   // (768, 768)
    const float* bout  = (const float*)d_in[3];   // (768,)
    (void)in_sizes; (void)n_in; (void)ws_size; (void)out_size;

    float* out      = (float*)d_out;                          // (8,1569,768)
    float* cls_attn = out + (long)B_ * NTOT * DIM;            // (96,1,1569)

    // Workspace: q/k/v head-major f16 + f16 attention output (b,t,h*d)
    const size_t qkvElems = (size_t)BH * NTOT * DIM_HEAD;     // 9,639,936
    _Float16* qbuf = (_Float16*)d_ws;
    _Float16* kbuf = qbuf + qkvElems;
    _Float16* vbuf = kbuf + qkvElems;
    _Float16* attnout = vbuf + qkvElems;                      // 12552 x 768 f16

    dim3 g1((M_ROWS + 63) / 64, QKV_N / 64);                  // 197 x 36
    wmma_gemm_kernel<0, float><<<g1, 128, 0, stream>>>(
        x, Wqkv, M_ROWS, DIM, QKV_N, qbuf, kbuf, vbuf, nullptr, nullptr);

    cls_attn_kernel<<<BH, 256, 0, stream>>>(qbuf, kbuf, vbuf, cls_attn, attnout);

    frame_attn_kernel<<<BH * F_, 256, 0, stream>>>(qbuf, kbuf, vbuf, attnout);

    dim3 g2((M_ROWS + 63) / 64, DIM / 64);                    // 197 x 12
    wmma_gemm_kernel<1, _Float16><<<g2, 128, 0, stream>>>(
        attnout, Wout, M_ROWS, DIM, DIM, nullptr, nullptr, nullptr, out, bout);
}
